// Attention_85186381349290
// MI455X (gfx1250) — matrix operand order repaired and checked by reading, not compiled, not run on hardware
//
#include <hip/hip_runtime.h>
#include <stdint.h>

#define B_SZ 256
#define P_SZ 196
#define E_SZ 2048
#define A_SZ 512
#define W_SZ 512

typedef __attribute__((ext_vector_type(16))) __bf16 bf16x16;
typedef __attribute__((ext_vector_type(8)))  float  f32x8;
typedef __attribute__((ext_vector_type(4)))  float  f32x4;
typedef __attribute__((ext_vector_type(4)))  unsigned int u32x4;
typedef __attribute__((ext_vector_type(2)))  unsigned int u32x2;

union Frag16 { bf16x16 v; u32x4 q[2]; };

__device__ __forceinline__ unsigned short f2bf(float f) {
  unsigned u = __float_as_uint(f);
  u += 0x7FFFu + ((u >> 16) & 1u);   // round-to-nearest-even
  return (unsigned short)(u >> 16);
}

// ---------------------------------------------------------------------------
// Prep 1: W_enc [E=2048][A=512] f32  ->  transposed bf16 [A=512][E=2048]
// so a WMMA B-fragment lane reads K as two contiguous 16-byte runs.
// ---------------------------------------------------------------------------
__global__ void prep_w_kernel(const float* __restrict__ W,
                              unsigned short* __restrict__ Wt) {
  int idx = blockIdx.x * 256 + threadIdx.x;   // 0 .. E*A-1 (coalesced read)
  int e = idx >> 9;          // / 512
  int a = idx & (A_SZ - 1);
  Wt[(size_t)a * E_SZ + e] = f2bf(W[idx]);
}

// ---------------------------------------------------------------------------
// Prep 2: bias[b][a] = b_enc[a] + b_dec[a] + decoder[b] . W_dec[:,a]
// ---------------------------------------------------------------------------
__global__ void prep_bias_kernel(const float* __restrict__ dec,
                                 const float* __restrict__ Wd,
                                 const float* __restrict__ be,
                                 const float* __restrict__ bd,
                                 float* __restrict__ bias) {
  int b = blockIdx.x;
  const float* dRow = dec + (size_t)b * W_SZ;
  for (int a = threadIdx.x; a < A_SZ; a += blockDim.x) {
    float acc = be[a] + bd[a];
    for (int k = 0; k < W_SZ; ++k)
      acc = fmaf(dRow[k], Wd[(size_t)k * A_SZ + a], acc);
    bias[(size_t)b * A_SZ + a] = acc;
  }
}

// ---------------------------------------------------------------------------
// Main fused attention kernel: one workgroup per batch (256 thr = 8 waves).
//  Phase 1: per 32-row super-tile (2 M-tiles), bf16 WMMA GEMM vs W_enc^T.
//           Each B-fragment feeds TWO WMMAs (one per M-tile) -> halves
//           L2 operand traffic per matrix op. Fused bias+relu+W_fin reduce.
//  Phase 2: in-block softmax over P=196.
//  Phase 3: weighted sum over encoder slice (L2-hot) -> out[b, 0..2047].
// ---------------------------------------------------------------------------
__global__ void __launch_bounds__(256) att_main_kernel(
    const float* __restrict__ enc, const float* __restrict__ bias,
    const unsigned short* __restrict__ Wt, const float* __restrict__ Wfin,
    const float* __restrict__ bfin, float* __restrict__ out)
{
  __shared__ unsigned short ldsA[32 * E_SZ];   // 128 KB bf16 A super-tile
  __shared__ float attScore[224];
  __shared__ float wts[224];
  __shared__ float sInvSum;

  const int b    = blockIdx.x;
  const int tid  = threadIdx.x;
  const int lane = tid & 31;
  const int wv   = tid >> 5;     // 8 waves
  const int rA   = lane & 15;    // row (A) / column (B) within tile
  const int kh   = lane >> 4;    // K-half selector
  const int nb   = wv * 64;      // this wave's N base (4 tiles of 16)

  const float* encB  = enc  + (size_t)b * P_SZ * E_SZ;
  const float* biasB = bias + (size_t)b * A_SZ;

  const float bf0 = bfin[0];     // shift-invariant under softmax, kept for fidelity
  for (int m = tid; m < 224; m += 256) attScore[m] = bf0;

  for (int mt2 = 0; mt2 < 7; ++mt2) {          // 7 * 32 = 224 >= 196 rows
    __syncthreads();
    const int rowBase = mt2 * 32;
    // ---- stage A super-tile: 32 x 2048 f32 -> bf16 LDS (zero-pad rows >=196)
    #pragma unroll 4
    for (int it = 0; it < 64; ++it) {
      int idx4 = tid + 256 * it;               // float4 index, 16384 total
      int e0   = idx4 << 2;
      int row  = e0 >> 11;
      int col  = e0 & (E_SZ - 1);
      f32x4 v = {0.f, 0.f, 0.f, 0.f};
      int gr = rowBase + row;
      if (gr < P_SZ)
        v = *(const f32x4*)(encB + (size_t)gr * E_SZ + col);
      u32x2 p;
      p.x = (unsigned)f2bf(v.x) | ((unsigned)f2bf(v.y) << 16);
      p.y = (unsigned)f2bf(v.z) | ((unsigned)f2bf(v.w) << 16);
      *(u32x2*)(ldsA + e0) = p;
    }
    __syncthreads();

    f32x8 acc0[4] = { {0,0,0,0,0,0,0,0}, {0,0,0,0,0,0,0,0},
                      {0,0,0,0,0,0,0,0}, {0,0,0,0,0,0,0,0} };
    f32x8 acc1[4] = { {0,0,0,0,0,0,0,0}, {0,0,0,0,0,0,0,0},
                      {0,0,0,0,0,0,0,0}, {0,0,0,0,0,0,0,0} };

    // A-fragment layout: lane<16 -> K {k0..k0+7, k0+16..k0+23}; lane>=16 -> +8
    const unsigned short* aRow0 = ldsA + rA * E_SZ + kh * 8;          // rows 0..15
    const unsigned short* aRow1 = ldsA + (16 + rA) * E_SZ + kh * 8;   // rows 16..31
    #pragma unroll 2
    for (int k0 = 0; k0 < E_SZ; k0 += 32) {
      Frag16 a0, a1;
      a0.q[0] = *(const u32x4*)(aRow0 + k0);
      a0.q[1] = *(const u32x4*)(aRow0 + k0 + 16);
      a1.q[0] = *(const u32x4*)(aRow1 + k0);
      a1.q[1] = *(const u32x4*)(aRow1 + k0 + 16);
      #pragma unroll
      for (int t = 0; t < 4; ++t) {
        // B-fragment: lane = column, K order matches A: {k0+8h..+7, k0+16+8h..+7}.
        // Loaded ONCE, consumed by two WMMAs (M-tile 0 and 1).
        const unsigned short* bp =
            Wt + (size_t)(nb + t * 16 + rA) * E_SZ + k0 + kh * 8;
        Frag16 bfr;
        bfr.q[0] = *(const u32x4*)bp;
        bfr.q[1] = *(const u32x4*)(bp + 16);
        acc0[t] = __builtin_amdgcn_wmma_f32_16x16x32_bf16(
            false, a0.v, false, bfr.v, (short)0, acc0[t], false, false);
        acc1[t] = __builtin_amdgcn_wmma_f32_16x16x32_bf16(
            false, a1.v, false, bfr.v, (short)0, acc1[t], false, false);
      }
    }

    // ---- epilogue: relu(acc + bias) * W_fin, reduce over this wave's 64 cols
    float s0[8] = {0,0,0,0,0,0,0,0};
    float s1[8] = {0,0,0,0,0,0,0,0};
    #pragma unroll
    for (int t = 0; t < 4; ++t) {
      int n = nb + t * 16 + rA;
      float db = biasB[n];
      float wf = Wfin[n];
      #pragma unroll
      for (int r = 0; r < 8; ++r) {
        float h0 = fmaxf(acc0[t][r] + db, 0.0f);
        s0[r] = fmaf(h0, wf, s0[r]);
        float h1 = fmaxf(acc1[t][r] + db, 0.0f);
        s1[r] = fmaf(h1, wf, s1[r]);
      }
    }
    #pragma unroll
    for (int r = 0; r < 8; ++r) {          // reduce across the 16-lane half
      float v0 = s0[r];
      v0 += __shfl_xor(v0, 1, 32);
      v0 += __shfl_xor(v0, 2, 32);
      v0 += __shfl_xor(v0, 4, 32);
      v0 += __shfl_xor(v0, 8, 32);
      s0[r] = v0;
      float v1 = s1[r];
      v1 += __shfl_xor(v1, 1, 32);
      v1 += __shfl_xor(v1, 2, 32);
      v1 += __shfl_xor(v1, 4, 32);
      v1 += __shfl_xor(v1, 8, 32);
      s1[r] = v1;
    }
    if (rA == 0) {                          // lane 0 -> M=r, lane 16 -> M=8+r
      int mg = rowBase + kh * 8;
      #pragma unroll
      for (int r = 0; r < 8; ++r) atomicAdd(&attScore[mg + r], s0[r]);
      #pragma unroll
      for (int r = 0; r < 8; ++r) atomicAdd(&attScore[mg + 16 + r], s1[r]);
    }
  }
  __syncthreads();

  // ---- phase 2: softmax over P=196 (wave 0)
  if (wv == 0) {
    float m = -3.0e38f;
    for (int p = lane; p < P_SZ; p += 32) m = fmaxf(m, attScore[p]);
    m = fmaxf(m, __shfl_xor(m, 1, 32));
    m = fmaxf(m, __shfl_xor(m, 2, 32));
    m = fmaxf(m, __shfl_xor(m, 4, 32));
    m = fmaxf(m, __shfl_xor(m, 8, 32));
    m = fmaxf(m, __shfl_xor(m, 16, 32));
    float ssum = 0.f;
    for (int p = lane; p < P_SZ; p += 32) {
      float e = __expf(attScore[p] - m);
      wts[p] = e;
      ssum += e;
    }
    ssum += __shfl_xor(ssum, 1, 32);
    ssum += __shfl_xor(ssum, 2, 32);
    ssum += __shfl_xor(ssum, 4, 32);
    ssum += __shfl_xor(ssum, 8, 32);
    ssum += __shfl_xor(ssum, 16, 32);
    if (lane == 0) sInvSum = 1.0f / ssum;
  }
  __syncthreads();

  // ---- phase 3: out[b,e] = (1/sum) * sum_p wts[p] * enc[b,p,e]
  const float inv = sInvSum;
  float o[8] = {0,0,0,0,0,0,0,0};
  for (int p = 0; p < P_SZ; ++p) {
    float wp = wts[p];
    const float* row = encB + (size_t)p * E_SZ;
    #pragma unroll
    for (int i = 0; i < 8; ++i)
      o[i] = fmaf(wp, row[tid + 256 * i], o[i]);
  }
  float* outB = out + (size_t)b * E_SZ;
  #pragma unroll
  for (int i = 0; i < 8; ++i) outB[tid + 256 * i] = o[i] * inv;
}

// ---------------------------------------------------------------------------
extern "C" void kernel_launch(void* const* d_in, const int* in_sizes, int n_in,
                              void* d_out, int out_size, void* d_ws, size_t ws_size,
                              hipStream_t stream) {
  const float* enc   = (const float*)d_in[0];
  const float* dec   = (const float*)d_in[1];
  const float* W_enc = (const float*)d_in[2];
  const float* b_enc = (const float*)d_in[3];
  const float* W_dec = (const float*)d_in[4];
  const float* b_dec = (const float*)d_in[5];
  const float* W_fin = (const float*)d_in[6];
  const float* b_fin = (const float*)d_in[7];
  float* out = (float*)d_out;

  char* ws = (char*)d_ws;
  unsigned short* Wt = (unsigned short*)ws;               // 2 MB bf16 W_enc^T
  float* bias = (float*)(ws + (size_t)E_SZ * A_SZ * 2);   // 512 KB fused bias

  prep_w_kernel<<<dim3((E_SZ * A_SZ) / 256), dim3(256), 0, stream>>>(W_enc, Wt);
  prep_bias_kernel<<<dim3(B_SZ), dim3(256), 0, stream>>>(dec, W_dec, b_enc, b_dec, bias);
  att_main_kernel<<<dim3(B_SZ), dim3(256), 0, stream>>>(enc, bias, Wt, W_fin, b_fin, out);
}
